// MixtureOfExperts_60172491817297
// MI455X (gfx1250) — matrix operand order repaired and checked by reading, not compiled, not run on hardware
//
#include <hip/hip_runtime.h>

#define D_IN   1024
#define D_HID  4096
#define D_OUT  1024
#define N_EXP  8
#define N_TOK  4096

typedef __attribute__((ext_vector_type(16))) __bf16 v16bf;
typedef __attribute__((ext_vector_type(8)))  __bf16 v8bf;
typedef __attribute__((ext_vector_type(8)))  float  v8f;

__device__ __forceinline__ v16bf cat16(v8bf lo, v8bf hi) {
    return __builtin_shufflevector(lo, hi, 0, 1, 2, 3, 4, 5, 6, 7,
                                            8, 9, 10, 11, 12, 13, 14, 15);
}

// Async global->LDS copy of 16 bytes per lane (ASYNCcnt-tracked, ISA 08 §4).
__device__ __forceinline__ void async_copy_b128(unsigned lds_off, const void* gaddr) {
    asm volatile("global_load_async_to_lds_b128 %0, %1, off"
                 :: "v"(lds_off), "v"(gaddr) : "memory");
}
__device__ __forceinline__ void wait_async0() {
    asm volatile("s_wait_asynccnt 0x0" ::: "memory");
}

// ---------------------------------------------------------------------------
// Kernel 0: zero output accumulator + per-expert counters.
// ---------------------------------------------------------------------------
__global__ void moe_zero(float* __restrict__ out, int* __restrict__ cnt, int total) {
    int i = blockIdx.x * blockDim.x + threadIdx.x;
    if (i < total) out[i] = 0.0f;
    if (blockIdx.x == 0 && threadIdx.x < N_EXP) cnt[threadIdx.x] = 0;
}

// ---------------------------------------------------------------------------
// Kernel 1: x fp32 -> bf16 (contiguous, coalesced).
// ---------------------------------------------------------------------------
__global__ void cvt_x(const float* __restrict__ src, __bf16* __restrict__ dst, int total) {
    int i = blockIdx.x * blockDim.x + threadIdx.x;
    if (i < total) dst[i] = (__bf16)src[i];
}

// ---------------------------------------------------------------------------
// Kernel 2: per-expert tiled transpose + fp32->bf16.  src[R][C] -> dst[C][R].
// 32x32 tiles through LDS so both the read and the write are coalesced.
// ---------------------------------------------------------------------------
__global__ __launch_bounds__(256)
void transpose_cvt(const float* __restrict__ src, __bf16* __restrict__ dst, int R, int C) {
    __shared__ float tile[32][33];
    const int bx = blockIdx.x * 32;             // C-dim base
    const int by = blockIdx.y * 32;             // R-dim base
    const float* s = src + (size_t)blockIdx.z * R * C;
    __bf16*      d = dst + (size_t)blockIdx.z * R * C;
    const int tx = threadIdx.x & 31;
    const int ty = threadIdx.x >> 5;            // 8 rows of 32
#pragma unroll
    for (int i = 0; i < 32; i += 8)
        tile[ty + i][tx] = s[(size_t)(by + ty + i) * C + bx + tx];
    __syncthreads();
#pragma unroll
    for (int i = 0; i < 32; i += 8)
        d[(size_t)(bx + ty + i) * R + by + tx] = (__bf16)tile[tx][ty + i];
}

// ---------------------------------------------------------------------------
// Kernel 3: gating. One wave32 per token; softmax -> top-2 -> renorm softmax,
// scatter (token, weight) into per-expert buckets.
// ---------------------------------------------------------------------------
__global__ __launch_bounds__(256)
void moe_gate(const float* __restrict__ x, const float* __restrict__ Wg,
              const float* __restrict__ bg,
              int* __restrict__ cnt, int* __restrict__ etok, float* __restrict__ ew) {
    const int wave = threadIdx.x >> 5;
    const int lane = threadIdx.x & 31;
    const int t = blockIdx.x * 8 + wave;
    if (t >= N_TOK) return;

    float g[N_EXP];
#pragma unroll
    for (int e = 0; e < N_EXP; ++e) g[e] = 0.0f;

    for (int d = lane; d < D_IN; d += 32) {
        float xv = x[(size_t)t * D_IN + d];
#pragma unroll
        for (int e = 0; e < N_EXP; ++e) g[e] += xv * Wg[d * N_EXP + e];
    }
#pragma unroll
    for (int off = 16; off > 0; off >>= 1) {
#pragma unroll
        for (int e = 0; e < N_EXP; ++e) g[e] += __shfl_xor(g[e], off, 32);
    }

    if (lane == 0) {
#pragma unroll
        for (int e = 0; e < N_EXP; ++e) g[e] += bg[e];
        float m = g[0];
#pragma unroll
        for (int e = 1; e < N_EXP; ++e) m = fmaxf(m, g[e]);
        float p[N_EXP], s = 0.0f;
#pragma unroll
        for (int e = 0; e < N_EXP; ++e) { p[e] = __expf(g[e] - m); s += p[e]; }
        float inv = 1.0f / s;
#pragma unroll
        for (int e = 0; e < N_EXP; ++e) p[e] *= inv;
        int i0 = 0;
#pragma unroll
        for (int e = 1; e < N_EXP; ++e) if (p[e] > p[i0]) i0 = e;
        int i1 = (i0 == 0) ? 1 : 0;
#pragma unroll
        for (int e = 0; e < N_EXP; ++e) if (e != i0 && p[e] > p[i1]) i1 = e;
        float w0 = 1.0f / (1.0f + __expf(p[i1] - p[i0]));   // p[i0] >= p[i1]
        float w1 = 1.0f - w0;

        int s0 = atomicAdd(&cnt[i0], 1);
        etok[i0 * N_TOK + s0] = t;
        ew  [i0 * N_TOK + s0] = w0;
        int s1 = atomicAdd(&cnt[i1], 1);
        etok[i1 * N_TOK + s1] = t;
        ew  [i1 * N_TOK + s1] = w1;
    }
}

// ---------------------------------------------------------------------------
// Kernel 4: fused expert MLP. One WG = 16 tokens of one expert, 8 wave32s.
// x rows async-staged to LDS; per-WMMA: A = 2x ds_load_b128 from LDS,
// B = 32 contiguous bytes of K-major bf16 weights (2x global_load_b128).
// GEMM1 -> bias+ReLU -> bf16 LDS chunk -> GEMM2 accumulated in registers.
// ---------------------------------------------------------------------------
__global__ __launch_bounds__(256)
void moe_mlp(const __bf16* __restrict__ xbf,
             const __bf16* __restrict__ W1T, const float* __restrict__ b1,
             const __bf16* __restrict__ W2T, const float* __restrict__ b2,
             const int* __restrict__ cnt, const int* __restrict__ etok,
             const float* __restrict__ ew, float* __restrict__ out) {
    const int e = blockIdx.y;
    const int n = cnt[e];
    const int tile = blockIdx.x;
    if (tile * 16 >= n) return;             // uniform exit: whole WG

    __shared__ __bf16 xA[16][D_IN];         // 32 KB token tile (A operand pool)
    __shared__ __bf16 hB[16][128];          // 4 KB hidden chunk
    __shared__ int    tok16[16];
    __shared__ float  gw16[16];

    const int tid  = threadIdx.x;
    const int lane = tid & 31;
    const int wave = tid >> 5;
    const int half = lane >> 4;             // 0: lanes 0-15, 1: lanes 16-31
    const int nn   = lane & 15;
    const int kbase = half * 8;             // K offset of this lane half is 2*kbase = 16*half

    if (tid < 16) {
        int slot  = tile * 16 + tid;
        int valid = slot < n;
        int s2    = valid ? slot : 0;
        tok16[tid] = etok[e * N_TOK + s2];
        gw16[tid]  = valid ? ew[e * N_TOK + s2] : 0.0f;
    }
    __syncthreads();

    // Async-stage the 16 gathered bf16 x-rows into LDS (16B per lane per issue).
    {
        unsigned ldsbase = (unsigned)(unsigned long long)(void*)&xA[0][0];
        for (int c = tid; c < 16 * (D_IN / 8); c += 256) {
            int r = c >> 7, o = (c & 127) * 8;
            const __bf16* g = xbf + (size_t)tok16[r] * D_IN + o;
            async_copy_b128(ldsbase + (unsigned)((r * D_IN + o) * 2), g);
        }
        wait_async0();
    }
    __syncthreads();

    const __bf16* W1Te = W1T + (size_t)e * D_IN * D_HID;   // [D_HID][D_IN] K-major
    const __bf16* W2Te = W2T + (size_t)e * D_HID * D_OUT;  // [D_OUT][D_HID] K-major
    const float*  b1e  = b1 + e * D_HID;
    const float*  b2e  = b2 + e * D_OUT;

    v8f yacc[8];
#pragma unroll
    for (int t = 0; t < 8; ++t) { v8f z = {}; yacc[t] = z; }

    for (int hc = 0; hc < D_HID / 128; ++hc) {
        // ---- GEMM1: this wave produces h[:, hcol0 .. hcol0+15] for this chunk
        const int hcol0 = hc * 128 + wave * 16;
        const __bf16* aRow = &xA[nn][0];
        const __bf16* bRow = W1Te + (size_t)(hcol0 + nn) * D_IN + kbase * 2;
        v8f hacc = {};
#pragma unroll 4
        for (int k0 = 0; k0 < D_IN; k0 += 32) {
            v8bf a0 = *(const v8bf*)(aRow + k0 + 2 * kbase);
            v8bf a1 = *(const v8bf*)(aRow + k0 + 8 + 2 * kbase);
            v16bf a = cat16(a0, a1);
            v16bf bm = *(const v16bf*)(bRow + k0);          // 32B contiguous
            __builtin_prefetch(bRow + k0 + 256, 0, 0);
            hacc = __builtin_amdgcn_wmma_f32_16x16x32_bf16(
                false, a, false, bm, (short)0, hacc, false, false);
        }
        // bias + ReLU -> bf16 chunk in LDS (C layout: M = r + 8*half, N = nn)
        float b1v = b1e[hcol0 + nn];
#pragma unroll
        for (int r = 0; r < 8; ++r) {
            float hv = hacc[r] + b1v;
            hv = hv > 0.0f ? hv : 0.0f;
            hB[r + half * 8][wave * 16 + nn] = (__bf16)hv;
        }
        __syncthreads();

        // ---- GEMM2: wave owns output columns [wave*128, wave*128+128)
#pragma unroll
        for (int t = 0; t < 8; ++t) {
            const int ncol = wave * 128 + t * 16;
            const __bf16* bRow2 = W2Te + (size_t)(ncol + nn) * D_HID
                                 + hc * 128 + kbase * 2;
#pragma unroll
            for (int k0 = 0; k0 < 128; k0 += 32) {
                const __bf16* row = &hB[nn][0];
                v8bf a0 = *(const v8bf*)(row + k0 + 2 * kbase);
                v8bf a1 = *(const v8bf*)(row + k0 + 8 + 2 * kbase);
                v16bf a = cat16(a0, a1);
                v16bf bm = *(const v16bf*)(bRow2 + k0);     // 32B contiguous
                yacc[t] = __builtin_amdgcn_wmma_f32_16x16x32_bf16(
                    false, a, false, bm, (short)0, yacc[t], false, false);
            }
        }
        __syncthreads();
    }

    // Epilogue: +b2, scale by gate weight, accumulate. Exactly two expert
    // contributions per (token, col) on a zeroed buffer -> order-independent.
#pragma unroll
    for (int t = 0; t < 8; ++t) {
        const int ncol = wave * 128 + t * 16;
        float b2v = b2e[ncol + nn];
#pragma unroll
        for (int r = 0; r < 8; ++r) {
            int m = r + half * 8;
            float v = gw16[m] * (yacc[t][r] + b2v);
            atomicAdd(&out[(size_t)tok16[m] * D_OUT + ncol + nn], v);
        }
    }
}

// ---------------------------------------------------------------------------
extern "C" void kernel_launch(void* const* d_in, const int* in_sizes, int n_in,
                              void* d_out, int out_size, void* d_ws, size_t ws_size,
                              hipStream_t stream) {
    const float* x  = (const float*)d_in[0];
    const float* W1 = (const float*)d_in[1];
    const float* b1 = (const float*)d_in[2];
    const float* W2 = (const float*)d_in[3];
    const float* b2 = (const float*)d_in[4];
    const float* Wg = (const float*)d_in[5];
    const float* bg = (const float*)d_in[6];
    // d_in[7] = top_k (== 2, baked into routing)
    float* out = (float*)d_out;

    // Workspace layout (~136.5 MB):
    //   [256B cnt][128KB etok][128KB ew][8MB x_bf16][64MB W1T_bf16][64MB W2T_bf16]
    char* ws = (char*)d_ws;
    size_t off = 0;
    int*    cnt  = (int*)(ws + off);    off += 256;
    int*    etok = (int*)(ws + off);    off += (size_t)N_EXP * N_TOK * sizeof(int);
    float*  ew   = (float*)(ws + off);  off += (size_t)N_EXP * N_TOK * sizeof(float);
    __bf16* xbf  = (__bf16*)(ws + off); off += (size_t)N_TOK * D_IN * 2;
    __bf16* W1T  = (__bf16*)(ws + off); off += (size_t)N_EXP * D_IN * D_HID * 2;
    __bf16* W2T  = (__bf16*)(ws + off); off += (size_t)N_EXP * D_HID * D_OUT * 2;

    moe_zero<<<(N_TOK * D_OUT + 255) / 256, 256, 0, stream>>>(out, cnt, N_TOK * D_OUT);
    cvt_x<<<(N_TOK * D_IN + 255) / 256, 256, 0, stream>>>(x, xbf, N_TOK * D_IN);
    {   // W1 [E][K=1024][H=4096] -> W1T [E][H][K] bf16
        dim3 g(D_HID / 32, D_IN / 32, N_EXP);
        transpose_cvt<<<g, 256, 0, stream>>>(W1, W1T, D_IN, D_HID);
    }
    {   // W2 [E][H=4096][O=1024] -> W2T [E][O][H] bf16
        dim3 g(D_OUT / 32, D_HID / 32, N_EXP);
        transpose_cvt<<<g, 256, 0, stream>>>(W2, W2T, D_HID, D_OUT);
    }
    moe_gate<<<N_TOK / 8, 256, 0, stream>>>(x, Wg, bg, cnt, etok, ew);
    dim3 grid(N_TOK / 16, N_EXP);
    moe_mlp<<<grid, 256, 0, stream>>>(xbf, W1T, b1, W2T, b2, cnt, etok, ew, out);
}
